// Cryformer_16269336117306
// MI455X (gfx1250) — matrix operand order repaired and checked by reading, not compiled, not run on hardware
//
#include <hip/hip_runtime.h>

typedef __attribute__((ext_vector_type(16))) __bf16 v16bf;
typedef __attribute__((ext_vector_type(8)))  __bf16 v8bf;
typedef __attribute__((ext_vector_type(8)))  float  v8f;
typedef __attribute__((ext_vector_type(4)))  unsigned int u32x4;
typedef __attribute__((ext_vector_type(8)))  int i32x8;
typedef __attribute__((ext_vector_type(4)))  int i32x4;

union bfrag {
  v16bf v;
  v8bf  h[2];
};

namespace cfg {
constexpr int N  = 1024, K = 20, D = 256, H = 4, F = 48, C = 32, S = 1024, SF = 1280;
constexpr int HF = H * F;    // 192
constexpr int NK = N * K;    // 20480
constexpr int TD = 3 * D;    // 768
constexpr int CD = C * D;    // 8192
constexpr int FH = F / 2;    // 24
constexpr float SQ2       = 1.41421356237309515f;
constexpr float INV_SCALE = 0.14433756729740643f;  // 1/sqrt(F)
}

// ---------------------------------------------------------------------------
// TDM 2D tile load: global (row-major, 2-byte elems, stride_d0 elems/row)
// -> LDS (rows packed contiguously, tile_d0 elems each).
// tensor_d0/d1 = REMAINING tensor extent from the tile origin, so the TDM
// zero-fills out-of-range k / rows for us (ISA: OOB reads return zero).
// Descriptor packing per CDNA5 ISA section 8.3-8.5 (D# groups 0/1; groups 2/3
// zero => 2D tensor, tile_dim2=0). This toolchain's builtin takes 6 args
// (g0, g1, g2, g3, second g1, cpol).
// ---------------------------------------------------------------------------
__device__ __forceinline__ void tdm_load_2d_bf16(
    unsigned lds_addr, const void* gaddr,
    unsigned tensor_d0, unsigned tensor_d1,
    unsigned tile_d0, unsigned tile_d1,
    unsigned stride_d0)
{
  unsigned long long ga = (unsigned long long)(uintptr_t)gaddr;
  u32x4 g0;
  g0[0] = 1u;                                            // count=1, user D#
  g0[1] = lds_addr;                                      // LDS byte address
  g0[2] = (unsigned)(ga & 0xffffffffu);                  // global addr [31:0]
  g0[3] = (unsigned)((ga >> 32) & 0x01ffffffu) | 0x80000000u; // [56:32] | type=2
  i32x8 g1;
  g1[0] = (int)(1u << 16);                               // data_size=1 -> 2 bytes
  g1[1] = (int)((tensor_d0 & 0xffffu) << 16);            // tensor_dim0[15:0]
  g1[2] = (int)((tensor_d0 >> 16) | ((tensor_d1 & 0xffffu) << 16));
  g1[3] = (int)((tensor_d1 >> 16) | (tile_d0 << 16));    // tile_dim0
  g1[4] = (int)(tile_d1 & 0xffffu);                      // tile_dim1, tile_dim2=0
  g1[5] = (int)stride_d0;                                // dim0 stride [31:0]
  g1[6] = 0;                                             // stride[47:32], dim1_stride lo
  g1[7] = 0;
  i32x4 z4 = {0, 0, 0, 0};
  i32x8 z8 = {0, 0, 0, 0, 0, 0, 0, 0};
  __builtin_amdgcn_tensor_load_to_lds(g0, g1, z4, z4, z8, 0);
}

// ---------------------------------------------------------------------------
// Generic bf16 WMMA GEMM (direct-global B):
//   C[M,Nd] = act( alpha * A[M,Kd] @ Bt[Nd,Kd]^T + bias )
// A row-major bf16 (lda), Bt row-major bf16 (ldb) == B transposed.
// Out-of-range M rows / Nd cols are CLAMPED (not masked): a bad A-row/B-col
// only pollutes C entries the store masks. Only K needs zero-fill (tail step).
// Each wave: 32(M) x 64(N) strip; block = 8 waves -> 128 x 128 tile.
// Used for small-K GEMMs (Kd<128 or Kd%32!=0).
// ---------------------------------------------------------------------------
__global__ __launch_bounds__(256) void k_gemm_bf16(
    const __bf16* __restrict__ A, int lda,
    const __bf16* __restrict__ Bt, int ldb,
    float* __restrict__ Cp, long ldc, long cstride,
    const float* __restrict__ bias,
    int M, int Nd, int Kd, float alpha, int act)
{
  const int lane = threadIdx.x & 31;
  const int wave = threadIdx.x >> 5;
  const int wm = wave & 3;
  const int wn = wave >> 2;
  const int m0 = blockIdx.y * 128 + wm * 32;
  const int n0 = blockIdx.x * 128 + wn * 64;
  if (m0 >= M || n0 >= Nd) return;  // wave-uniform early out

  const int l16 = lane & 15;
  const int lh  = lane >> 4;

  const __bf16* arow0 = A + (size_t)min(m0 + l16,      M - 1) * lda;
  const __bf16* arow1 = A + (size_t)min(m0 + 16 + l16, M - 1) * lda;
  const __bf16* brow0 = Bt + (size_t)min(n0 +  0 + l16, Nd - 1) * ldb;
  const __bf16* brow1 = Bt + (size_t)min(n0 + 16 + l16, Nd - 1) * ldb;
  const __bf16* brow2 = Bt + (size_t)min(n0 + 32 + l16, Nd - 1) * ldb;
  const __bf16* brow3 = Bt + (size_t)min(n0 + 48 + l16, Nd - 1) * ldb;

  // A (16x32 bf16): elem e   -> k = 16*lh+e
  // B (32x16 bf16): elem e   -> k = 16*lh+e
  const int aoff = 16 * lh;
  const int boff = 16 * lh;

  v8f acc[2][4];
#pragma unroll
  for (int i = 0; i < 2; ++i)
#pragma unroll
    for (int t = 0; t < 4; ++t) acc[i][t] = (v8f){};

  const int Kmain = Kd & ~31;
  for (int k0 = 0; k0 < Kmain; k0 += 32) {
    bfrag a0, a1, b0, b1, b2, b3;
    a0.h[0] = *(const v8bf*)(arow0 + k0 + aoff);
    a0.h[1] = *(const v8bf*)(arow0 + k0 + aoff + 8);
    a1.h[0] = *(const v8bf*)(arow1 + k0 + aoff);
    a1.h[1] = *(const v8bf*)(arow1 + k0 + aoff + 8);
    b0.h[0] = *(const v8bf*)(brow0 + k0 + boff);
    b0.h[1] = *(const v8bf*)(brow0 + k0 + boff + 8);
    b1.h[0] = *(const v8bf*)(brow1 + k0 + boff);
    b1.h[1] = *(const v8bf*)(brow1 + k0 + boff + 8);
    b2.h[0] = *(const v8bf*)(brow2 + k0 + boff);
    b2.h[1] = *(const v8bf*)(brow2 + k0 + boff + 8);
    b3.h[0] = *(const v8bf*)(brow3 + k0 + boff);
    b3.h[1] = *(const v8bf*)(brow3 + k0 + boff + 8);
    __builtin_prefetch(arow0 + k0 + 32 + aoff, 0, 1);
    __builtin_prefetch(arow1 + k0 + 32 + aoff, 0, 1);
    __builtin_prefetch(brow0 + k0 + 32 + boff, 0, 1);
    __builtin_prefetch(brow2 + k0 + 32 + boff, 0, 1);

    acc[0][0] = __builtin_amdgcn_wmma_f32_16x16x32_bf16(false, a0.v, false, b0.v, (short)0, acc[0][0], false, false);
    acc[0][1] = __builtin_amdgcn_wmma_f32_16x16x32_bf16(false, a0.v, false, b1.v, (short)0, acc[0][1], false, false);
    acc[0][2] = __builtin_amdgcn_wmma_f32_16x16x32_bf16(false, a0.v, false, b2.v, (short)0, acc[0][2], false, false);
    acc[0][3] = __builtin_amdgcn_wmma_f32_16x16x32_bf16(false, a0.v, false, b3.v, (short)0, acc[0][3], false, false);
    acc[1][0] = __builtin_amdgcn_wmma_f32_16x16x32_bf16(false, a1.v, false, b0.v, (short)0, acc[1][0], false, false);
    acc[1][1] = __builtin_amdgcn_wmma_f32_16x16x32_bf16(false, a1.v, false, b1.v, (short)0, acc[1][1], false, false);
    acc[1][2] = __builtin_amdgcn_wmma_f32_16x16x32_bf16(false, a1.v, false, b2.v, (short)0, acc[1][2], false, false);
    acc[1][3] = __builtin_amdgcn_wmma_f32_16x16x32_bf16(false, a1.v, false, b3.v, (short)0, acc[1][3], false, false);
  }

  if (Kmain < Kd) {  // single zero-filled tail step (only Kd=48 hits this)
    const int k0 = Kmain;
    bfrag a0, a1;
#pragma unroll
    for (int e = 0; e < 8; ++e) {
      int k1 = k0 + aoff + e;
      int k2 = k0 + 8 + aoff + e;
      a0.v[e]     = (k1 < Kd) ? arow0[k1] : (__bf16)0.0f;
      a0.v[8 + e] = (k2 < Kd) ? arow0[k2] : (__bf16)0.0f;
      a1.v[e]     = (k1 < Kd) ? arow1[k1] : (__bf16)0.0f;
      a1.v[8 + e] = (k2 < Kd) ? arow1[k2] : (__bf16)0.0f;
    }
    const __bf16* brows[4] = {brow0, brow1, brow2, brow3};
#pragma unroll
    for (int t = 0; t < 4; ++t) {
      bfrag b;
#pragma unroll
      for (int e = 0; e < 16; ++e) {
        int kk = k0 + boff + e;
        b.v[e] = (kk < Kd) ? brows[t][kk] : (__bf16)0.0f;
      }
      acc[0][t] = __builtin_amdgcn_wmma_f32_16x16x32_bf16(false, a0.v, false, b.v, (short)0, acc[0][t], false, false);
      acc[1][t] = __builtin_amdgcn_wmma_f32_16x16x32_bf16(false, a1.v, false, b.v, (short)0, acc[1][t], false, false);
    }
  }

#pragma unroll
  for (int i = 0; i < 2; ++i) {
    const int rbase = m0 + i * 16 + (lh << 3);
#pragma unroll
    for (int t = 0; t < 4; ++t) {
      int col = n0 + t * 16 + l16;
      if (col >= Nd) continue;
      float bv = bias ? bias[col] : 0.0f;
#pragma unroll
      for (int j = 0; j < 8; ++j) {
        int row = rbase + j;
        if (row < M) {
          float v = acc[i][t][j] * alpha + bv;
          if (act == 1)      v = 1.0f / (1.0f + __expf(-v));
          else if (act == 2) v = v / (1.0f + __expf(-v));
          Cp[(size_t)row * ldc + (size_t)col * cstride] = v;
        }
      }
    }
  }
}

// ---------------------------------------------------------------------------
// TDM-staged WMMA GEMM (requires Kd % 32 == 0; used for Kd >= 128).
// The block's 128-column B slab is staged into LDS by the Tensor Data Mover
// (one 2D descriptor per 128-element k-slab, 32 KB), giving 4x reuse of B
// from LDS instead of 4 redundant L2 fetches. OOB k / rows are zero-filled by
// the TDM, so no guards are needed on the LDS side.
// ---------------------------------------------------------------------------
#define KSLAB 128
__global__ __launch_bounds__(256) void k_gemm_bf16_tdm(
    const __bf16* __restrict__ A, int lda,
    const __bf16* __restrict__ Bt, int ldb,
    float* __restrict__ Cp, long ldc, long cstride,
    const float* __restrict__ bias,
    int M, int Nd, int Kd, float alpha, int act)
{
  __shared__ __bf16 Bs[128 * KSLAB];   // [col_local][k_local], 32 KB

  const int lane = threadIdx.x & 31;
  const int wave = threadIdx.x >> 5;
  const int wm = wave & 3;
  const int wn = wave >> 2;
  const int m0  = blockIdx.y * 128 + wm * 32;
  const int n0b = blockIdx.x * 128;          // block column base (shared slab)
  const int n0  = n0b + wn * 64;
  // no early return: every wave participates in the barriers.

  const int l16 = lane & 15;
  const int lh  = lane >> 4;
  const int aoff = 16 * lh;
  const int boff = 16 * lh;

  const __bf16* arow0 = A + (size_t)min(m0 + l16,      M - 1) * lda;
  const __bf16* arow1 = A + (size_t)min(m0 + 16 + l16, M - 1) * lda;

  const unsigned lds_base = (unsigned)(uintptr_t)(void*)&Bs[0];

  v8f acc[2][4];
#pragma unroll
  for (int i = 0; i < 2; ++i)
#pragma unroll
    for (int t = 0; t < 4; ++t) acc[i][t] = (v8f){};

  for (int k0 = 0; k0 < Kd; k0 += KSLAB) {
    __syncthreads();  // slab buffer free
    if (wave == 0) {
      tdm_load_2d_bf16(lds_base,
                       Bt + (size_t)n0b * ldb + k0,
                       (unsigned)(Kd - k0),          // remaining k  (zero-fill past)
                       (unsigned)(Nd - n0b),         // remaining B rows (cols of C)
                       KSLAB, 128, (unsigned)ldb);
      __builtin_amdgcn_s_wait_tensorcnt(0);
    }
    __syncthreads();  // slab visible to all waves

#pragma unroll
    for (int ks = 0; ks < KSLAB / 32; ++ks) {
      const int kk = k0 + ks * 32;
      if (kk >= Kd) break;  // uniform; trailing zero-filled steps skipped
      bfrag a0, a1, b0, b1, b2, b3;
      a0.h[0] = *(const v8bf*)(arow0 + kk + aoff);
      a0.h[1] = *(const v8bf*)(arow0 + kk + aoff + 8);
      a1.h[0] = *(const v8bf*)(arow1 + kk + aoff);
      a1.h[1] = *(const v8bf*)(arow1 + kk + aoff + 8);
      const __bf16* bs0 = Bs + (size_t)(wn * 64 +  0 + l16) * KSLAB + ks * 32 + boff;
      const __bf16* bs1 = Bs + (size_t)(wn * 64 + 16 + l16) * KSLAB + ks * 32 + boff;
      const __bf16* bs2 = Bs + (size_t)(wn * 64 + 32 + l16) * KSLAB + ks * 32 + boff;
      const __bf16* bs3 = Bs + (size_t)(wn * 64 + 48 + l16) * KSLAB + ks * 32 + boff;
      b0.h[0] = *(const v8bf*)(bs0);  b0.h[1] = *(const v8bf*)(bs0 + 8);
      b1.h[0] = *(const v8bf*)(bs1);  b1.h[1] = *(const v8bf*)(bs1 + 8);
      b2.h[0] = *(const v8bf*)(bs2);  b2.h[1] = *(const v8bf*)(bs2 + 8);
      b3.h[0] = *(const v8bf*)(bs3);  b3.h[1] = *(const v8bf*)(bs3 + 8);
      __builtin_prefetch(arow0 + kk + 32 + aoff, 0, 1);
      __builtin_prefetch(arow1 + kk + 32 + aoff, 0, 1);

      acc[0][0] = __builtin_amdgcn_wmma_f32_16x16x32_bf16(false, a0.v, false, b0.v, (short)0, acc[0][0], false, false);
      acc[0][1] = __builtin_amdgcn_wmma_f32_16x16x32_bf16(false, a0.v, false, b1.v, (short)0, acc[0][1], false, false);
      acc[0][2] = __builtin_amdgcn_wmma_f32_16x16x32_bf16(false, a0.v, false, b2.v, (short)0, acc[0][2], false, false);
      acc[0][3] = __builtin_amdgcn_wmma_f32_16x16x32_bf16(false, a0.v, false, b3.v, (short)0, acc[0][3], false, false);
      acc[1][0] = __builtin_amdgcn_wmma_f32_16x16x32_bf16(false, a1.v, false, b0.v, (short)0, acc[1][0], false, false);
      acc[1][1] = __builtin_amdgcn_wmma_f32_16x16x32_bf16(false, a1.v, false, b1.v, (short)0, acc[1][1], false, false);
      acc[1][2] = __builtin_amdgcn_wmma_f32_16x16x32_bf16(false, a1.v, false, b2.v, (short)0, acc[1][2], false, false);
      acc[1][3] = __builtin_amdgcn_wmma_f32_16x16x32_bf16(false, a1.v, false, b3.v, (short)0, acc[1][3], false, false);
    }
  }

#pragma unroll
  for (int i = 0; i < 2; ++i) {
    const int rbase = m0 + i * 16 + (lh << 3);
#pragma unroll
    for (int t = 0; t < 4; ++t) {
      int col = n0 + t * 16 + l16;
      if (col >= Nd) continue;
      float bv = bias ? bias[col] : 0.0f;
#pragma unroll
      for (int j = 0; j < 8; ++j) {
        int row = rbase + j;
        if (row < M) {
          float v = acc[i][t][j] * alpha + bv;
          if (act == 1)      v = 1.0f / (1.0f + __expf(-v));
          else if (act == 2) v = v / (1.0f + __expf(-v));
          Cp[(size_t)row * ldc + (size_t)col * cstride] = v;
        }
      }
    }
  }
}

// ---------------------------------------------------------------------------
// Elementwise / helper kernels
// ---------------------------------------------------------------------------
__global__ void k_transpose_bf16(const float* __restrict__ in, int ldin,
                                 int R, int Cc, __bf16* __restrict__ out, int ldout)
{
  long i = (long)blockIdx.x * blockDim.x + threadIdx.x;
  long tot = (long)R * Cc;
  if (i >= tot) return;
  int r = (int)(i / Cc), c = (int)(i % Cc);
  out[(size_t)c * ldout + r] = (__bf16)in[(size_t)r * ldin + c];
}

__global__ void k_f32_to_bf16(const float* __restrict__ in, int ldin,
                              int rows, int cols, __bf16* __restrict__ out, int ldout)
{
  long i = (long)blockIdx.x * blockDim.x + threadIdx.x;
  long tot = (long)rows * cols;
  if (i >= tot) return;
  int r = (int)(i / cols), c = (int)(i % cols);
  out[(size_t)r * ldout + c] = (__bf16)in[(size_t)r * ldin + c];
}

__global__ void k_mul_bf16(const float* __restrict__ a, const float* __restrict__ b,
                           __bf16* __restrict__ out, long n)
{
  long i = (long)blockIdx.x * blockDim.x + threadIdx.x;
  if (i >= n) return;
  out[i] = (__bf16)(a[i] * b[i]);
}

// LayerNorm over last dim (Dlen == blockDim.x == 256), optional residual,
// optional f32 and bf16 outputs.
__global__ __launch_bounds__(256) void k_ln(
    const float* __restrict__ x, const float* __restrict__ res, float rs,
    const float* __restrict__ g, const float* __restrict__ b,
    float* __restrict__ outf, __bf16* __restrict__ outbf, int Dlen)
{
  __shared__ float red[256];
  size_t row = blockIdx.x;
  int j = threadIdx.x;
  float v = x[row * Dlen + j];
  if (res) v += rs * res[row * Dlen + j];
  red[j] = v;
  __syncthreads();
  for (int s = 128; s > 0; s >>= 1) { if (j < s) red[j] += red[j + s]; __syncthreads(); }
  float mean = red[0] / (float)Dlen;
  __syncthreads();
  float xc = v - mean;
  red[j] = xc * xc;
  __syncthreads();
  for (int s = 128; s > 0; s >>= 1) { if (j < s) red[j] += red[j + s]; __syncthreads(); }
  float var = red[0] / (float)Dlen;
  float y = xc * rsqrtf(var + 1e-5f) * g[j] + b[j];
  if (outf)  outf[row * Dlen + j] = y;
  if (outbf) outbf[row * Dlen + j] = (__bf16)y;
}

// Sequence softmax over s with mask. scores laid out (n,s,h); writes bf16
// probabilities in (h,n,s) layout (GEMM A-operand layout).
__global__ __launch_bounds__(256) void k_softmax_sa(
    const float* __restrict__ sc, const unsigned char* __restrict__ mask,
    __bf16* __restrict__ w)
{
  using namespace cfg;
  int n = blockIdx.x / H, h = blockIdx.x % H;
  __shared__ float red[256];
  float vals[4];
  float vmax = -1e30f;
#pragma unroll
  for (int t = 0; t < 4; ++t) {
    int s = threadIdx.x + t * 256;
    float v = sc[((size_t)n * S + s) * H + h];
    v = mask[s] ? v : -1e9f;
    vals[t] = v;
    vmax = fmaxf(vmax, v);
  }
  red[threadIdx.x] = vmax;
  __syncthreads();
  for (int s = 128; s > 0; s >>= 1) { if (threadIdx.x < s) red[threadIdx.x] = fmaxf(red[threadIdx.x], red[threadIdx.x + s]); __syncthreads(); }
  vmax = red[0];
  __syncthreads();
  float sum = 0.0f;
#pragma unroll
  for (int t = 0; t < 4; ++t) { vals[t] = __expf(vals[t] - vmax); sum += vals[t]; }
  red[threadIdx.x] = sum;
  __syncthreads();
  for (int s = 128; s > 0; s >>= 1) { if (threadIdx.x < s) red[threadIdx.x] += red[threadIdx.x + s]; __syncthreads(); }
  float inv = 1.0f / red[0];
#pragma unroll
  for (int t = 0; t < 4; ++t) {
    int s = threadIdx.x + t * 256;
    w[(size_t)h * N * S + (size_t)n * S + s] = (__bf16)(vals[t] * inv);
  }
}

// RoPE: out[n, h*F+f] = in*cos(pos[n, f%24]) + rot_half(in)*sin(...)
__global__ void k_rope(const float* __restrict__ in, const float* __restrict__ pos,
                       float* __restrict__ out)
{
  using namespace cfg;
  long i = (long)blockIdx.x * blockDim.x + threadIdx.x;
  if (i >= (long)N * HF) return;
  int n = (int)(i / HF);
  int r = (int)(i % HF);
  int f = r % F;
  float ang = pos[(size_t)n * FH + (f % FH)];
  float c = __cosf(ang), s = __sinf(ang);
  float t = in[i];
  float rot = (f < FH) ? -in[i + FH] : in[i - FH];
  out[i] = t * c + rot * s;
}

// Neighbor-attention scores: s[n,kk,h] = <qr[n,h], kr[e,h]>/SCALE + bias2[n,kk,h]
__global__ void k_na_score(const float* __restrict__ qr, const float* __restrict__ kr,
                           const float* __restrict__ bias2, const int* __restrict__ edge,
                           float* __restrict__ sout)
{
  using namespace cfg;
  long i = (long)blockIdx.x * blockDim.x + threadIdx.x;
  if (i >= (long)N * K * H) return;
  int h = (int)(i % H);
  int nk = (int)(i / H);
  int n = nk / K;
  int e = edge[nk];
  const float* qp = qr + (size_t)n * HF + h * F;
  const float* kp = kr + (size_t)e * HF + h * F;
  float s = 0.0f;
#pragma unroll
  for (int f = 0; f < F; ++f) s += qp[f] * kp[f];
  sout[i] = s * INV_SCALE + bias2[(size_t)nk * H + h];
}

__global__ void k_na_softmax(const float* __restrict__ s, float* __restrict__ w)
{
  using namespace cfg;
  long i = (long)blockIdx.x * blockDim.x + threadIdx.x;
  if (i >= (long)N * H) return;
  int n = (int)(i / H), h = (int)(i % H);
  float m = -1e30f;
#pragma unroll
  for (int kk = 0; kk < K; ++kk) m = fmaxf(m, s[((size_t)n * K + kk) * H + h]);
  float sum = 0.0f;
  float e_[K];
#pragma unroll
  for (int kk = 0; kk < K; ++kk) { e_[kk] = __expf(s[((size_t)n * K + kk) * H + h] - m); sum += e_[kk]; }
  float inv = 1.0f / sum;
#pragma unroll
  for (int kk = 0; kk < K; ++kk) w[((size_t)n * K + kk) * H + h] = e_[kk] * inv;
}

// o[n,h,f] = gate * sum_kk w[n,kk,h]*v[e(n,kk), h*F+f]  -> bf16 (N,HF)
__global__ void k_na_out(const float* __restrict__ w, const float* __restrict__ vy,
                         const float* __restrict__ gate, const int* __restrict__ edge,
                         __bf16* __restrict__ out)
{
  using namespace cfg;
  long i = (long)blockIdx.x * blockDim.x + threadIdx.x;
  if (i >= (long)N * HF) return;
  int n = (int)(i / HF);
  int r = (int)(i % HF);
  int h = r / F;
  float acc = 0.0f;
#pragma unroll
  for (int kk = 0; kk < K; ++kk) {
    int e = edge[(size_t)n * K + kk];
    acc += w[((size_t)n * K + kk) * H + h] * vy[(size_t)e * HF + r];
  }
  out[i] = (__bf16)(gate[i] * acc);
}

// Outer-product contraction: out[nk,d] = op_b2[d] + sum_j b[e,j] * T[n, j*D+d]
__global__ void k_opout(const float* __restrict__ ab, const float* __restrict__ T,
                        const float* __restrict__ b2, const int* __restrict__ edge,
                        float* __restrict__ out)
{
  using namespace cfg;
  long i = (long)blockIdx.x * blockDim.x + threadIdx.x;
  if (i >= (long)NK * D) return;
  int nk = (int)(i / D);
  int d = (int)(i % D);
  int n = nk / K;
  int e = edge[nk];
  const float* zb = ab + (size_t)e * 64 + C;       // b half of ab row
  const float* Tp = T + (size_t)n * CD + d;
  float acc = b2[d];
#pragma unroll
  for (int j = 0; j < C; ++j) acc += zb[j] * Tp[(size_t)j * D];
  out[i] = acc;
}

// Edge attention: scores over 2K=40 keys, softmax -> wE[(nk*H+h)*40 + l2]
__global__ void k_ea_attn(const float* __restrict__ qE, const float* __restrict__ kE,
                          const float* __restrict__ biaE, const int* __restrict__ edge,
                          float* __restrict__ wE)
{
  using namespace cfg;
  long i = (long)blockIdx.x * blockDim.x + threadIdx.x;
  if (i >= (long)N * K * H) return;
  int h = (int)(i % H);
  int nk = (int)(i / H);
  int n = nk / K;
  int e = edge[nk];
  float q[F];
#pragma unroll
  for (int f = 0; f < F; ++f) q[f] = qE[(size_t)nk * HF + h * F + f];
  float sc[2 * K];
  float m = -1e30f;
  for (int l = 0; l < K; ++l) {
    const float* kp = kE + ((size_t)n * K + l) * HF + h * F;
    float s = 0.0f;
#pragma unroll
    for (int f = 0; f < F; ++f) s += q[f] * kp[f];
    s = s * INV_SCALE + biaE[((size_t)n * K + l) * H + h];
    sc[l] = s;
    m = fmaxf(m, s);
  }
  for (int l = 0; l < K; ++l) {
    const float* kp = kE + ((size_t)e * K + l) * HF + h * F;
    float s = 0.0f;
#pragma unroll
    for (int f = 0; f < F; ++f) s += q[f] * kp[f];
    s = s * INV_SCALE + biaE[((size_t)e * K + l) * H + h];
    sc[K + l] = s;
    m = fmaxf(m, s);
  }
  float sum = 0.0f;
#pragma unroll
  for (int l = 0; l < 2 * K; ++l) { sc[l] = __expf(sc[l] - m); sum += sc[l]; }
  float inv = 1.0f / sum;
  float* wp = wE + (size_t)i * (2 * K);
#pragma unroll
  for (int l = 0; l < 2 * K; ++l) wp[l] = sc[l] * inv;
}

// o[nk, h*F+f] = gate * ( sum_l w_self*vE[n*K+l] + sum_l w_nb*vE[e*K+l] ) -> bf16
__global__ void k_ea_out(const float* __restrict__ wE, const float* __restrict__ vE,
                         const float* __restrict__ gateE, const int* __restrict__ edge,
                         __bf16* __restrict__ out)
{
  using namespace cfg;
  long i = (long)blockIdx.x * blockDim.x + threadIdx.x;
  if (i >= (long)NK * HF) return;
  int nk = (int)(i / HF);
  int r = (int)(i % HF);
  int h = r / F;
  int n = nk / K;
  int e = edge[nk];
  const float* wp = wE + ((size_t)nk * H + h) * (2 * K);
  float acc = 0.0f;
#pragma unroll
  for (int l = 0; l < K; ++l) acc += wp[l] * vE[((size_t)n * K + l) * HF + r];
#pragma unroll
  for (int l = 0; l < K; ++l) acc += wp[K + l] * vE[((size_t)e * K + l) * HF + r];
  out[i] = (__bf16)(gateE[i] * acc);
}

// ---------------------------------------------------------------------------
// Host orchestration
// ---------------------------------------------------------------------------
extern "C" void kernel_launch(void* const* d_in, const int* in_sizes, int n_in,
                              void* d_out, int out_size, void* d_ws, size_t ws_size,
                              hipStream_t stream)
{
  using namespace cfg;
  (void)in_sizes; (void)n_in; (void)out_size; (void)ws_size;

  // inputs (setup_inputs dict order, params pytree-flattened in insertion order)
  const float* x1        = (const float*)d_in[0];
  const float* x2        = (const float*)d_in[1];
  const float* pos       = (const float*)d_in[2];
  const float* pse       = (const float*)d_in[3];
  const float* p_sa_q    = (const float*)d_in[4];
  const float* p_sa_k    = (const float*)d_in[5];
  const float* p_sa_v    = (const float*)d_in[6];
  const float* p_sa_gw   = (const float*)d_in[7];
  const float* p_sa_gb   = (const float*)d_in[8];
  const float* p_sa_bw   = (const float*)d_in[9];
  const float* p_sa_bb   = (const float*)d_in[10];
  const float* p_sa_ng   = (const float*)d_in[11];
  const float* p_sa_nb   = (const float*)d_in[12];
  const float* p_na_q    = (const float*)d_in[13];
  const float* p_na_k    = (const float*)d_in[14];
  const float* p_na_v    = (const float*)d_in[15];
  const float* p_na_blng = (const float*)d_in[16];
  const float* p_na_blnb = (const float*)d_in[17];
  const float* p_na_bw   = (const float*)d_in[18];
  const float* p_na_gw   = (const float*)d_in[19];
  const float* p_na_gb   = (const float*)d_in[20];
  const float* p_na_bkw  = (const float*)d_in[21];
  const float* p_na_bkb  = (const float*)d_in[22];
  const float* p_na_ng   = (const float*)d_in[23];
  const float* p_na_nb   = (const float*)d_in[24];
  const float* p_t1_w1   = (const float*)d_in[25];
  const float* p_t1_w2   = (const float*)d_in[26];
  const float* p_t1_w3   = (const float*)d_in[27];
  const float* p_t1_ng   = (const float*)d_in[28];
  const float* p_t1_nb   = (const float*)d_in[29];
  const float* p_op_w    = (const float*)d_in[30];
  const float* p_op_b    = (const float*)d_in[31];
  const float* p_op_w2   = (const float*)d_in[32];
  const float* p_op_b2   = (const float*)d_in[33];
  const float* p_on_g    = (const float*)d_in[34];
  const float* p_on_b    = (const float*)d_in[35];
  const float* p_ea_q    = (const float*)d_in[36];
  const float* p_ea_k    = (const float*)d_in[37];
  const float* p_ea_v    = (const float*)d_in[38];
  const float* p_ea_bia  = (const float*)d_in[39];
  const float* p_ea_gw   = (const float*)d_in[40];
  const float* p_ea_gb   = (const float*)d_in[41];
  const float* p_ea_bkw  = (const float*)d_in[42];
  const float* p_ea_bkb  = (const float*)d_in[43];
  const float* p_ea_ng   = (const float*)d_in[44];
  const float* p_ea_nb   = (const float*)d_in[45];
  const float* p_t2_w1   = (const float*)d_in[46];
  const float* p_t2_w2   = (const float*)d_in[47];
  const float* p_t2_w3   = (const float*)d_in[48];
  const float* p_t2_ng   = (const float*)d_in[49];
  const float* p_t2_nb   = (const float*)d_in[50];
  const int*   edge      = (const int*)d_in[51];
  const unsigned char* mask = (const unsigned char*)d_in[52];

  float* outy1   = (float*)d_out;                       // (N,D)
  float* outy2   = outy1 + (size_t)N * D;               // (N,K,D)
  float* outattn = outy2 + (size_t)NK * D;              // (N,S,H)

  // workspace bump allocator (256B aligned)
  char* wsp = (char*)d_ws;
  size_t off = 0;
  auto alloc = [&](size_t bytes) -> void* {
    off = (off + 255) & ~(size_t)255;
    void* p = wsp + off;
    off += bytes;
    return p;
  };
  auto fb = [&](size_t n) { return (float*)alloc(n * sizeof(float)); };
  auto bb = [&](size_t n) { return (__bf16*)alloc(n * sizeof(__bf16)); };

  // bf16 transposed weights
  __bf16 *sa_qT = bb((size_t)HF * D), *sa_kT = bb((size_t)HF * SF), *sa_vT = bb((size_t)HF * SF);
  __bf16 *sa_gT = bb((size_t)HF * D), *sa_bT = bb((size_t)D * HF);
  __bf16 *na_qT = bb((size_t)HF * D), *na_kT = bb((size_t)HF * D), *na_vT = bb((size_t)HF * D);
  __bf16 *na_biT = bb((size_t)H * D), *na_gT = bb((size_t)HF * D), *na_bT = bb((size_t)D * HF);
  __bf16 *t1_w1T = bb((size_t)TD * D), *t1_w2T = bb((size_t)TD * D), *t1_w3T = bb((size_t)D * TD);
  __bf16 *op_wT = bb((size_t)64 * D), *op_w2T = bb((size_t)CD * C);
  __bf16 *ea_qT = bb((size_t)HF * D), *ea_kT = bb((size_t)HF * D), *ea_vT = bb((size_t)HF * D);
  __bf16 *ea_biT = bb((size_t)H * D), *ea_gT = bb((size_t)HF * D), *ea_bT = bb((size_t)D * HF);
  __bf16 *t2_w1T = bb((size_t)TD * D), *t2_w2T = bb((size_t)TD * D), *t2_w3T = bb((size_t)D * TD);

  // activations
  __bf16 *x1_bf = bb((size_t)N * D), *pse_bf = bb((size_t)S * SF);
  float  *q_sa = fb((size_t)N * HF);  __bf16 *q_bf = bb((size_t)N * HF);
  float  *k_sa = fb((size_t)S * HF);  __bf16 *k_bf = bb((size_t)S * HF);
  float  *v_sa = fb((size_t)S * HF);  __bf16 *vT_bf = bb((size_t)H * F * S);
  __bf16 *w_bf = bb((size_t)H * N * S);
  float  *attno = fb((size_t)N * HF);
  float  *gate  = fb((size_t)N * HF);
  __bf16 *og_bf = bb((size_t)N * HF);
  float  *t0    = fb((size_t)NK * D);            // reused back/transition output
  float  *y1f   = fb((size_t)N * D); __bf16 *y1_bf = bb((size_t)N * D);
  float  *qna = fb((size_t)N * HF), *kna = fb((size_t)N * HF), *vna = fb((size_t)N * HF);
  float  *qr = fb((size_t)N * HF), *kr = fb((size_t)N * HF);
  __bf16 *bln_bf = bb((size_t)NK * D);
  float  *bias2 = fb((size_t)NK * H);
  float  *sna = fb((size_t)N * K * H), *wna = fb((size_t)N * K * H);
  float  *g1 = fb((size_t)NK * TD), *g2 = fb((size_t)NK * TD);
  __bf16 *prod_bf = bb((size_t)NK * TD);
  float  *ab = fb((size_t)N * 64);  __bf16 *a_bf = bb((size_t)N * C);
  float  *Tb = fb((size_t)N * CD);
  float  *y2in = fb((size_t)NK * D); __bf16 *y2_bf = bb((size_t)NK * D);
  float  *qE = fb((size_t)NK * HF), *kE = fb((size_t)NK * HF), *vE = fb((size_t)NK * HF);
  float  *gE = fb((size_t)NK * HF), *biaE = fb((size_t)NK * H);
  float  *wEb = fb((size_t)NK * H * 2 * K);
  __bf16 *ogE_bf = bb((size_t)NK * HF);

  // launch helpers: TDM-staged kernel for large-K GEMMs, direct otherwise
  auto gemm = [&](const __bf16* A, int lda, const __bf16* Bt, int ldb,
                  float* Cp, long ldc, long cs, const float* bias,
                  int M, int Nd, int Kd, float alpha, int act) {
    dim3 g((Nd + 127) / 128, (M + 127) / 128);
    if (Kd >= 128 && (Kd % 32) == 0)
      k_gemm_bf16_tdm<<<g, 256, 0, stream>>>(A, lda, Bt, ldb, Cp, ldc, cs, bias, M, Nd, Kd, alpha, act);
    else
      k_gemm_bf16<<<g, 256, 0, stream>>>(A, lda, Bt, ldb, Cp, ldc, cs, bias, M, Nd, Kd, alpha, act);
  };
  auto tw = [&](const float* W, int Kd, int Nd, __bf16* Wt) {
    long tot = (long)Kd * Nd;
    k_transpose_bf16<<<dim3((unsigned)((tot + 255) / 256)), 256, 0, stream>>>(W, Nd, Kd, Nd, Wt, Kd);
  };
  auto conv = [&](const float* in, int ldin, int rows, int cols, __bf16* out, int ldout) {
    long tot = (long)rows * cols;
    k_f32_to_bf16<<<dim3((unsigned)((tot + 255) / 256)), 256, 0, stream>>>(in, ldin, rows, cols, out, ldout);
  };
  auto ln = [&](const float* x, const float* res, float rs, const float* g, const float* b,
                float* outf, __bf16* outbf, int rows) {
    k_ln<<<dim3((unsigned)rows), 256, 0, stream>>>(x, res, rs, g, b, outf, outbf, D);
  };

  // ---- weight conversion (transpose to (Nd x Kd) bf16) ----
  tw(p_sa_q, D, HF, sa_qT);   tw(p_sa_k, SF, HF, sa_kT);  tw(p_sa_v, SF, HF, sa_vT);
  tw(p_sa_gw, D, HF, sa_gT);  tw(p_sa_bw, HF, D, sa_bT);
  tw(p_na_q, D, HF, na_qT);   tw(p_na_k, D, HF, na_kT);   tw(p_na_v, D, HF, na_vT);
  tw(p_na_bw, D, H, na_biT);  tw(p_na_gw, D, HF, na_gT);  tw(p_na_bkw, HF, D, na_bT);
  tw(p_t1_w1, D, TD, t1_w1T); tw(p_t1_w2, D, TD, t1_w2T); tw(p_t1_w3, TD, D, t1_w3T);
  tw(p_op_w, D, 64, op_wT);   tw(p_op_w2, C, CD, op_w2T); // op_w2 viewed as (C x C*D)
  tw(p_ea_q, D, HF, ea_qT);   tw(p_ea_k, D, HF, ea_kT);   tw(p_ea_v, D, HF, ea_vT);
  tw(p_ea_bia, D, H, ea_biT); tw(p_ea_gw, D, HF, ea_gT);  tw(p_ea_bkw, HF, D, ea_bT);
  tw(p_t2_w1, D, TD, t2_w1T); tw(p_t2_w2, D, TD, t2_w2T); tw(p_t2_w3, TD, D, t2_w3T);

  // ---- stage 1: sequence attention ----
  conv(x1, D, N, D, x1_bf, D);
  conv(pse, SF, S, SF, pse_bf, SF);
  gemm(x1_bf, D, sa_qT, D, q_sa, HF, 1, nullptr, N, HF, D, 1.0f, 0);
  conv(q_sa, HF, N, HF, q_bf, HF);
  gemm(pse_bf, SF, sa_kT, SF, k_sa, HF, 1, nullptr, S, HF, SF, 1.0f, 0);
  conv(k_sa, HF, S, HF, k_bf, HF);
  gemm(pse_bf, SF, sa_vT, SF, v_sa, HF, 1, nullptr, S, HF, SF, 1.0f, 0);
  for (int h = 0; h < H; ++h) {
    k_transpose_bf16<<<dim3((S * F + 255) / 256), 256, 0, stream>>>(
        v_sa + h * F, HF, S, F, vT_bf + (size_t)h * F * S, S);
    // raw attn scores -> d_out (n,s,h) layout, alpha = 1/sqrt(F)
    gemm(q_bf + h * F, HF, k_bf + h * F, HF, outattn + h, (long)S * H, H,
         nullptr, N, S, F, INV_SCALE, 0);
  }
  k_softmax_sa<<<dim3(N * H), 256, 0, stream>>>(outattn, mask, w_bf);
  for (int h = 0; h < H; ++h)
    gemm(w_bf + (size_t)h * N * S, S, vT_bf + (size_t)h * F * S, S,
         attno + h * F, HF, 1, nullptr, N, F, S, 1.0f, 0);
  gemm(x1_bf, D, sa_gT, D, gate, HF, 1, p_sa_gb, N, HF, D, 1.0f, 1);
  k_mul_bf16<<<dim3((N * HF + 255) / 256), 256, 0, stream>>>(gate, attno, og_bf, (long)N * HF);
  gemm(og_bf, HF, sa_bT, HF, t0, D, 1, p_sa_bb, N, D, HF, 1.0f, 0);
  ln(t0, x1, SQ2, p_sa_ng, p_sa_nb, y1f, y1_bf, N);

  // ---- stage 2: neighbor attention ----
  gemm(y1_bf, D, na_qT, D, qna, HF, 1, nullptr, N, HF, D, 1.0f, 0);
  gemm(y1_bf, D, na_kT, D, kna, HF, 1, nullptr, N, HF, D, 1.0f, 0);
  gemm(y1_bf, D, na_vT, D, vna, HF, 1, nullptr, N, HF, D, 1.0f, 0);
  k_rope<<<dim3((N * HF + 255) / 256), 256, 0, stream>>>(qna, pos, qr);
  k_rope<<<dim3((N * HF + 255) / 256), 256, 0, stream>>>(kna, pos, kr);
  ln(x2, nullptr, 0.0f, p_na_blng, p_na_blnb, nullptr, bln_bf, NK);
  gemm(bln_bf, D, na_biT, D, bias2, H, 1, nullptr, NK, H, D, 1.0f, 0);
  k_na_score<<<dim3((N * K * H + 255) / 256), 256, 0, stream>>>(qr, kr, bias2, edge, sna);
  k_na_softmax<<<dim3((N * H + 255) / 256), 256, 0, stream>>>(sna, wna);
  gemm(y1_bf, D, na_gT, D, gate, HF, 1, p_na_gb, N, HF, D, 1.0f, 1);
  k_na_out<<<dim3((N * HF + 255) / 256), 256, 0, stream>>>(wna, vna, gate, edge, og_bf);
  gemm(og_bf, HF, na_bT, HF, t0, D, 1, p_na_bkb, N, D, HF, 1.0f, 0);
  ln(t0, y1f, SQ2, p_na_ng, p_na_nb, y1f, y1_bf, N);

  // transition 1
  gemm(y1_bf, D, t1_w1T, D, g1, TD, 1, nullptr, N, TD, D, 1.0f, 2);
  gemm(y1_bf, D, t1_w2T, D, g2, TD, 1, nullptr, N, TD, D, 1.0f, 0);
  k_mul_bf16<<<dim3((unsigned)(((long)N * TD + 255) / 256)), 256, 0, stream>>>(g1, g2, prod_bf, (long)N * TD);
  gemm(prod_bf, TD, t1_w3T, TD, t0, D, 1, nullptr, N, D, TD, 1.0f, 0);
  ln(t0, y1f, SQ2, p_t1_ng, p_t1_nb, outy1, y1_bf, N);   // y1 output

  // ---- stage 3: outer product ----
  gemm(y1_bf, D, op_wT, D, ab, 64, 1, p_op_b, N, 64, D, 1.0f, 0);
  conv(ab, 64, N, C, a_bf, C);
  gemm(a_bf, C, op_w2T, C, Tb, CD, 1, nullptr, N, CD, C, 1.0f, 0); // T[n] = a[n] @ W2'
  k_opout<<<dim3((unsigned)(((long)NK * D + 255) / 256)), 256, 0, stream>>>(ab, Tb, p_op_b2, edge, t0);
  ln(t0, x2, SQ2, p_on_g, p_on_b, y2in, y2_bf, NK);

  // ---- stage 4: edge attention ----
  gemm(y2_bf, D, ea_qT, D, qE, HF, 1, nullptr, NK, HF, D, 1.0f, 0);
  gemm(y2_bf, D, ea_kT, D, kE, HF, 1, nullptr, NK, HF, D, 1.0f, 0);
  gemm(y2_bf, D, ea_vT, D, vE, HF, 1, nullptr, NK, HF, D, 1.0f, 0);
  gemm(y2_bf, D, ea_gT, D, gE, HF, 1, p_ea_gb, NK, HF, D, 1.0f, 1);
  gemm(y2_bf, D, ea_biT, D, biaE, H, 1, nullptr, NK, H, D, 1.0f, 0);
  k_ea_attn<<<dim3((N * K * H + 255) / 256), 256, 0, stream>>>(qE, kE, biaE, edge, wEb);
  k_ea_out<<<dim3((unsigned)(((long)NK * HF + 255) / 256)), 256, 0, stream>>>(wEb, vE, gE, edge, ogE_bf);
  gemm(ogE_bf, HF, ea_bT, HF, t0, D, 1, p_ea_bkb, NK, D, HF, 1.0f, 0);
  ln(t0, y2in, SQ2, p_ea_ng, p_ea_nb, y2in, y2_bf, NK);

  // transition 2
  gemm(y2_bf, D, t2_w1T, D, g1, TD, 1, nullptr, NK, TD, D, 1.0f, 2);
  gemm(y2_bf, D, t2_w2T, D, g2, TD, 1, nullptr, NK, TD, D, 1.0f, 0);
  k_mul_bf16<<<dim3((unsigned)(((long)NK * TD + 255) / 256)), 256, 0, stream>>>(g1, g2, prod_bf, (long)NK * TD);
  gemm(prod_bf, TD, t2_w3T, TD, t0, D, 1, nullptr, NK, D, TD, 1.0f, 0);
  ln(t0, y2in, SQ2, p_t2_ng, p_t2_nb, outy2, nullptr, NK);  // y2 output
}
